// MoELayer_13039520711498
// MI455X (gfx1250) — matrix operand order repaired and checked by reading, not compiled, not run on hardware
//
#include <hip/hip_runtime.h>
#include <hip/hip_bf16.h>
#include <math.h>

typedef __attribute__((ext_vector_type(16))) __bf16 v16bf;
typedef __attribute__((ext_vector_type(8)))  __bf16 v8bf;
typedef __attribute__((ext_vector_type(8)))  float  v8f;

#define E_EXPERTS 8
#define D_MODEL   1024
#define F_FFN     4096
#define FC        128     // F-chunk processed per outer iteration
#define NWAVES    8

// ---- helpers ---------------------------------------------------------------

// A-fragment for V_WMMA_*_16x16x32_bf16 (ISA 7.12.2, 16-bit A 16x32):
// lane m (0..15): V0..3 = K k..k+7, V4..7 = K k+16..k+23
// lane m+16    : V0..3 = K k+8..k+15, V4..7 = K k+24..k+31
// caller passes k' = kb + 8*half; we read 8 elems at k' and 8 at k'+16.
__device__ __forceinline__ v16bf load_a_frag(const __bf16* rowbase, int k) {
  v8bf lo = *(const v8bf*)(rowbase + k);
  v8bf hi = *(const v8bf*)(rowbase + k + 16);
  v16bf a;
#pragma unroll
  for (int i = 0; i < 8; ++i) { a[i] = lo[i]; a[i + 8] = hi[i]; }
  return a;
}

__device__ __forceinline__ float gelu_exact(float v) {
  return 0.5f * v * (1.0f + erff(v * 0.70710678118654752f));
}

// ---- kernel 0: zero output + expert counters --------------------------------

__global__ void moe_zero_kernel(float* __restrict__ out, size_t n,
                                int* __restrict__ counts) {
  size_t i = (size_t)blockIdx.x * blockDim.x + threadIdx.x;
  if (i < E_EXPERTS) counts[i] = 0;
  size_t stride = (size_t)gridDim.x * blockDim.x;
  for (size_t k = i * 4; k + 3 < n; k += stride * 4)
    *(float4*)(out + k) = make_float4(0.f, 0.f, 0.f, 0.f);
}

// ---- kernel 1: router (one wave32 per token) --------------------------------

__global__ void moe_router_kernel(const float* __restrict__ x,
                                  const float* __restrict__ Wr,
                                  int* __restrict__ counts,
                                  int* __restrict__ tokidx,
                                  float* __restrict__ tokgate,
                                  int n_tokens) {
  int gid  = blockIdx.x * blockDim.x + threadIdx.x;
  int t    = gid >> 5;
  int lane = threadIdx.x & 31;
  if (t >= n_tokens) return;

  const float* xt = x + (size_t)t * D_MODEL;
  float acc[E_EXPERTS];
#pragma unroll
  for (int e = 0; e < E_EXPERTS; ++e) acc[e] = 0.f;

  for (int d = lane; d < D_MODEL; d += 32) {
    float xv = xt[d];
#pragma unroll
    for (int e = 0; e < E_EXPERTS; ++e) acc[e] += xv * Wr[e * D_MODEL + d];
  }
#pragma unroll
  for (int e = 0; e < E_EXPERTS; ++e) {
#pragma unroll
    for (int off = 16; off > 0; off >>= 1)
      acc[e] += __shfl_xor(acc[e], off, 32);
  }

  if (lane == 0) {
    int   i0 = 0;      float l0 = acc[0];
#pragma unroll
    for (int e = 1; e < E_EXPERTS; ++e) if (acc[e] > l0) { l0 = acc[e]; i0 = e; }
    int   i1 = (i0 == 0) ? 1 : 0;
    float l1 = -3.0e38f;
#pragma unroll
    for (int e = 0; e < E_EXPERTS; ++e)
      if (e != i0 && acc[e] > l1) { l1 = acc[e]; i1 = e; }

    // softmax over the two selected logits (l0 >= l1)
    float w  = __expf(l1 - l0);
    float g0 = 1.0f / (1.0f + w);
    float g1 = 1.0f - g0;

    int p0 = atomicAdd(&counts[i0], 1);
    tokidx [i0 * n_tokens + p0] = t;
    tokgate[i0 * n_tokens + p0] = g0;
    int p1 = atomicAdd(&counts[i1], 1);
    tokidx [i1 * n_tokens + p1] = t;
    tokgate[i1 * n_tokens + p1] = g1;
  }
}

// ---- kernel 2: per-expert fused FFN (x@W1 -> gelu -> @W2, gated scatter) ----

__global__ __launch_bounds__(256, 1)
void moe_expert_ffn_kernel(const float* __restrict__ x,
                           const float* __restrict__ W1,
                           const float* __restrict__ W2,
                           const int*   __restrict__ counts,
                           const int*   __restrict__ tokidx,
                           const float* __restrict__ tokgate,
                           float*       __restrict__ out,
                           int n_tokens) {
  __shared__ __bf16 sX[16][D_MODEL + 8];  // +8 bf16 pad: rows land on distinct banks
  __shared__ __bf16 sH[16][FC + 8];
  __shared__ int    sTok[16];
  __shared__ float  sG[16];

  const int e    = blockIdx.y;
  const int cnt  = counts[e];
  const int row0 = blockIdx.x * 16;
  if (row0 >= cnt) return;               // uniform early exit, no barrier crossed

  const int tid  = threadIdx.x;
  const int lane = tid & 31;
  const int wid  = tid >> 5;             // 8 waves
  const int hf   = lane >> 4;            // half-wave select (ISA fragment layouts)
  const int ln   = lane & 15;

  if (tid < 16) {
    int s = row0 + tid;
    if (s < cnt) { sTok[tid] = tokidx[e * n_tokens + s];
                   sG[tid]   = tokgate[e * n_tokens + s]; }
    else         { sTok[tid] = 0; sG[tid] = 0.f; }
  }
  __syncthreads();

  { // stage 16 gathered token rows, fp32 -> bf16
    const int  row   = tid >> 4;
    const int  c0    = (tid & 15) << 6;          // 64 cols per thread
    const bool valid = (row0 + row) < cnt;
    const float* xr  = x + (size_t)sTok[row] * D_MODEL + c0;
#pragma unroll 4
    for (int c = 0; c < 64; c += 4) {
      float4 v = valid ? *(const float4*)(xr + c) : make_float4(0.f, 0.f, 0.f, 0.f);
      sX[row][c0 + c + 0] = (__bf16)v.x;
      sX[row][c0 + c + 1] = (__bf16)v.y;
      sX[row][c0 + c + 2] = (__bf16)v.z;
      sX[row][c0 + c + 3] = (__bf16)v.w;
    }
  }
  __syncthreads();

  const float* __restrict__ W1e = W1 + (size_t)e * D_MODEL * F_FFN;
  const float* __restrict__ W2e = W2 + (size_t)e * F_FFN * D_MODEL;
  const __bf16* xrow = &sX[ln][0];
  const __bf16* hrow = &sH[ln][0];
  const int n0 = wid * (D_MODEL / NWAVES);       // 128 output cols per wave

  v8f acc[8];
#pragma unroll
  for (int t = 0; t < 8; ++t) acc[t] = (v8f)0.0f;

  for (int chunk = 0; chunk < F_FFN / FC; ++chunk) {
    // ---- phase 1: this wave's 16-wide h slice; K loop over full D -----------
    v8f hacc = (v8f)0.0f;
    const float* w1c = W1e + (size_t)(chunk * FC + wid * 16 + ln);
    for (int kb = 0; kb < D_MODEL; kb += 32) {
      v16bf a = load_a_frag(xrow, kb + 8 * hf);
      const float* wp = w1c + (size_t)(kb + 8 * hf) * F_FFN;
      if (kb + 32 < D_MODEL)
        __builtin_prefetch(wp + 32 * (size_t)F_FFN, 0, 0);  // global_prefetch next K-block
      v16bf b;                               // ISA B layout: lane = column, K packed
#pragma unroll
      for (int kk = 0; kk < 16; ++kk) b[kk] = (__bf16)wp[(size_t)(kk + (kk & 8)) * F_FFN];
      hacc = __builtin_amdgcn_wmma_f32_16x16x32_bf16(false, a, false, b,
                                                     (short)0, hacc, false, false);
    }
    __syncthreads();                         // prior chunk's sH readers done
#pragma unroll
    for (int r = 0; r < 8; ++r)              // C layout: row = r + 8*hf, col = ln
      sH[r + 8 * hf][wid * 16 + ln] = (__bf16)gelu_exact(hacc[r]);
    __syncthreads();

    // ---- phase 2: acc += h_chunk @ W2[chunk rows, n0:n0+128] ----------------
    v16bf afr[4];
#pragma unroll
    for (int ks = 0; ks < 4; ++ks) afr[ks] = load_a_frag(hrow, ks * 32 + 8 * hf);
#pragma unroll
    for (int t = 0; t < 8; ++t) {
      const float* w2c = W2e + (size_t)(n0 + t * 16 + ln);
#pragma unroll
      for (int ks = 0; ks < 4; ++ks) {
        const float* wp = w2c + (size_t)(chunk * FC + ks * 32 + 8 * hf) * D_MODEL;
        v16bf b;
#pragma unroll
        for (int kk = 0; kk < 16; ++kk) b[kk] = (__bf16)wp[(size_t)(kk + (kk & 8)) * D_MODEL];
        acc[t] = __builtin_amdgcn_wmma_f32_16x16x32_bf16(false, afr[ks], false, b,
                                                         (short)0, acc[t], false, false);
      }
    }
  }

  // ---- gated scatter-add (each token has 2 expert contributions) ------------
#pragma unroll
  for (int t = 0; t < 8; ++t) {
    const int col = n0 + t * 16 + ln;
#pragma unroll
    for (int r = 0; r < 8; ++r) {
      const int rr = r + 8 * hf;             // C layout row
      atomicAdd(out + (size_t)sTok[rr] * D_MODEL + col, acc[t][r] * sG[rr]);
    }
  }
}

// ---- launch -----------------------------------------------------------------

extern "C" void kernel_launch(void* const* d_in, const int* in_sizes, int n_in,
                              void* d_out, int out_size, void* d_ws, size_t ws_size,
                              hipStream_t stream) {
  const float* x  = (const float*)d_in[0];   // [B,S,D]
  const float* Wr = (const float*)d_in[1];   // [E,D]
  const float* W1 = (const float*)d_in[2];   // [E,D,F]
  const float* W2 = (const float*)d_in[3];   // [E,F,D]
  float* out = (float*)d_out;                // [B,S,D]

  const int n_tokens = in_sizes[0] / D_MODEL;  // B*S

  // workspace layout: [counts: 8 ints (pad 256B)] [tokidx: E*NT ints] [gates: E*NT floats]
  int*   counts  = (int*)d_ws;
  int*   tokidx  = (int*)((char*)d_ws + 256);
  float* tokgate = (float*)((char*)d_ws + 256 +
                            (size_t)E_EXPERTS * n_tokens * sizeof(int));

  moe_zero_kernel<<<1024, 256, 0, stream>>>(out, (size_t)n_tokens * D_MODEL, counts);

  int router_blocks = (n_tokens * 32 + 255) / 256;
  moe_router_kernel<<<router_blocks, 256, 0, stream>>>(x, Wr, counts, tokidx,
                                                       tokgate, n_tokens);

  dim3 grid((n_tokens + 15) / 16, E_EXPERTS);
  moe_expert_ffn_kernel<<<grid, 256, 0, stream>>>(x, W1, W2, counts, tokidx,
                                                  tokgate, out, n_tokens);
}
